// MultiHeadAttention_936302871321
// MI455X (gfx1250) — matrix operand order repaired and checked by reading, not compiled, not run on hardware
//
#include <hip/hip_runtime.h>
#include <stdint.h>

#define DMODEL 768
#define NHEADS 12
#define DHEAD  64
#define SEQ    1024
#define BATCH  8
#define TOK    (BATCH * SEQ)   // 8192 tokens

typedef __attribute__((ext_vector_type(16))) _Float16 v16h;
typedef __attribute__((ext_vector_type(8)))  float    v8f;

union Frag16 { v16h v; unsigned u[8]; };

__device__ __forceinline__ v8f wmma_f16(v16h a, v16h b, v8f c) {
    // D = A(16x32 f16) * B(32x16 f16) + C(16x16 f32)
    return __builtin_amdgcn_wmma_f32_16x16x32_f16(false, a, false, b, (short)0, c,
                                                  false, false);
}

// Async copy of 16 bytes/lane from global to LDS (gfx1250, tracked by ASYNCcnt).
__device__ __forceinline__ void async_copy_b128(unsigned lds_addr, unsigned long long gaddr) {
    asm volatile("global_load_async_to_lds_b128 %0, %1, off"
                 :: "v"(lds_addr), "v"(gaddr) : "memory");
}
__device__ __forceinline__ void async_wait0() {
    asm volatile("s_wait_asynccnt 0x0" ::: "memory");
}

// ---------------------------------------------------------------------------
// proj_gemm: Y = X @ W^T + bias  (torch nn.Linear convention, W is [O, K])
//   MODE 0: f16 out, layout [B, H, L, DHEAD]          (q, k)
//   MODE 1: f16 out, layout [B, H, DHEAD, L]          (v, transposed for PV)
//   MODE 2: f32 out, layout [TOK, DMODEL]             (final projection)
//   AF16: A operand is f16 (ctx buffer) vs f32 (raw inputs, cvt fused in load)
// One wave computes one 16x16 tile; 8 waves per 256-thread block.
// ---------------------------------------------------------------------------
template<int MODE, bool AF16>
__global__ __launch_bounds__(256) void proj_gemm(const void*  __restrict__ Xv,
                                                 const float* __restrict__ W,
                                                 const float* __restrict__ bias,
                                                 _Float16*    __restrict__ out16,
                                                 float*       __restrict__ out32)
{
    const int lane = threadIdx.x & 31;
    const int wave = threadIdx.x >> 5;
    const int tile = blockIdx.x * 8 + wave;        // 512 x 48 tiles
    const int mt   = tile / (DMODEL / 16);
    const int nt   = tile % (DMODEL / 16);
    const int h    = lane >> 4;
    const int n    = lane & 15;
    const int arow = mt * 16 + n;                  // token row for A fragment
    const int bcol = nt * 16 + n;                  // output feature for B fragment

    const float*    Xf = (const float*)Xv;
    const unsigned* Xh = (const unsigned*)Xv;

    v8f acc = {};
    for (int k0 = 0; k0 < DMODEL; k0 += 32) {
        Frag16 a, b;
#pragma unroll
        for (int g = 0; g < 8; ++g) {
            // A layout (16-bit, 16x32): lanes 0-15 K={0..7,16..23}, lanes 16-31 K={8..15,24..31}
            const int ka = k0 + ((g >> 2) << 4) + (h << 3) + ((g & 3) << 1);
            if (AF16) {
                a.u[g] = Xh[((size_t)arow * DMODEL + ka) >> 1];
            } else {
                const float* p = Xf + (size_t)arow * DMODEL + ka;
                a.v[2 * g]     = (_Float16)p[0];
                a.v[2 * g + 1] = (_Float16)p[1];
            }
            // B layout (16-bit, 32x16): same K order as A (lanes 0-15 K={0..7,16..23}, lanes 16-31 K={8..15,24..31})
            const int kb = k0 + ((g >> 2) << 4) + (h << 3) + ((g & 3) << 1);
            const float* q = W + (size_t)bcol * DMODEL + kb;
            b.v[2 * g]     = (_Float16)q[0];
            b.v[2 * g + 1] = (_Float16)q[1];
        }
        acc = wmma_f16(a.v, b.v, acc);
    }

    const float bv = bias[bcol];
#pragma unroll
    for (int r = 0; r < 8; ++r) {
        const int   m   = mt * 16 + r + 8 * h;     // C/D layout: row = r + 8*(lane>>4)
        const float val = acc[r] + bv;
        if (MODE == 2) {
            out32[(size_t)m * DMODEL + bcol] = val;
        } else {
            const int bb = m >> 10;                // batch
            const int sl = m & (SEQ - 1);          // seq pos
            const int hh = bcol >> 6;              // head
            const int d  = bcol & (DHEAD - 1);     // dim in head
            size_t idx;
            if (MODE == 0) idx = ((size_t)(bb * NHEADS + hh) * SEQ + sl) * DHEAD + d;
            else           idx = ((size_t)(bb * NHEADS + hh) * DHEAD + d) * SEQ + sl;
            out16[idx] = (_Float16)val;
        }
    }
}

// ---------------------------------------------------------------------------
// attn_kernel: flash attention. Block = 4 waves (128 thr); block handles one
// (b, h, 64 query rows); each wave owns 16 query rows. Online softmax, fp32
// accumulators, f16 WMMA. Per 32-key chunk, K (32x64 f16) and V^T (64x32 f16)
// are staged once per workgroup into LDS with async-to-LDS copies (shared by
// all 4 waves), then fragments are built with ds loads.
// ---------------------------------------------------------------------------
__global__ __launch_bounds__(128) void attn_kernel(const _Float16* __restrict__ q16,
                                                   const _Float16* __restrict__ k16,
                                                   const _Float16* __restrict__ vT16,
                                                   const unsigned char* __restrict__ mask,
                                                   _Float16* __restrict__ ctx)
{
    __shared__ __align__(16) _Float16 kbuf[32][DHEAD];   // [key_local][d]   4KB
    __shared__ __align__(16) _Float16 vbuf[DHEAD][32];   // [d][key_local]   4KB
    __shared__ __align__(16) _Float16 pbuf[4][16][32];   // per-wave P stage 4KB

    const int tid  = threadIdx.x;
    const int lane = tid & 31;
    const int wave = tid >> 5;
    const int h    = lane >> 4;
    const int n    = lane & 15;

    const int wg = blockIdx.x;            // B*H*(SEQ/64) = 1536
    const int qt = wg & 15;               // 16 q-tiles of 64 rows per (b,h)
    const int bh = wg >> 4;               // b*NHEADS + head
    const int b  = bh / NHEADS;
    const int hd = bh - b * NHEADS;
    const int q0 = qt * 64 + wave * 16;   // first query row of this wave (0..1023)

    const unsigned* qp = (const unsigned*)(q16 + (size_t)bh * SEQ * DHEAD);
    const unsigned char* mp = mask + ((size_t)b * SEQ + q0) * SEQ;

    // Byte bases for async staging
    const unsigned long long kgb = (unsigned long long)(uintptr_t)(k16  + (size_t)bh * SEQ * DHEAD);
    const unsigned long long vgb = (unsigned long long)(uintptr_t)(vT16 + (size_t)bh * DHEAD * SEQ);
    const unsigned ldsk = (unsigned)(uintptr_t)&kbuf[0][0];
    const unsigned ldsv = (unsigned)(uintptr_t)&vbuf[0][0];
    // Each thread owns 2 x 16B segments of K and 2 x 16B segments of V per chunk.
    const int s0 = tid * 2, s1 = tid * 2 + 1;          // 0..255 (x16B = 4KB)

    const unsigned* kbu = (const unsigned*)&kbuf[0][0];
    const unsigned* vbu = (const unsigned*)&vbuf[0][0];

    // Q A-fragments: rows q0..q0+15, K-dim = head dim (two chunks of 32)
    Frag16 qa0, qa1;
#pragma unroll
    for (int g = 0; g < 8; ++g) {
        const int kk = ((g >> 2) << 4) + (h << 3) + ((g & 3) << 1);
        qa0.u[g] = qp[(((size_t)(q0 + n)) * DHEAD + kk) >> 1];
        qa1.u[g] = qp[(((size_t)(q0 + n)) * DHEAD + 32 + kk) >> 1];
    }

    v8f o0 = {}, o1 = {}, o2 = {}, o3 = {};       // 16 x 64 fp32 output accum
    float mrow[8], srow[8];
#pragma unroll
    for (int r = 0; r < 8; ++r) { mrow[r] = -__builtin_inff(); srow[r] = 0.f; }

    unsigned* pw = (unsigned*)&pbuf[wave][0][0];

    for (int kc = 0; kc < SEQ; kc += 32) {
        // --- stage K chunk (contiguous 4KB) and V^T chunk into LDS (async) ---
        // K: keys kc..kc+31, rows of 128B; segment s covers key s>>3, dwords (s&7)*4
        async_copy_b128(ldsk + s0 * 16, kgb + (size_t)kc * (DHEAD * 2) + s0 * 16);
        async_copy_b128(ldsk + s1 * 16, kgb + (size_t)kc * (DHEAD * 2) + s1 * 16);
        // V^T: d-row s>>2, 16B part s&3 of the 64B run vT[d][kc..kc+31]
        async_copy_b128(ldsv + s0 * 16,
                        vgb + (size_t)(s0 >> 2) * (SEQ * 2) + (size_t)kc * 2 + (s0 & 3) * 16);
        async_copy_b128(ldsv + s1 * 16,
                        vgb + (size_t)(s1 >> 2) * (SEQ * 2) + (size_t)kc * 2 + (s1 & 3) * 16);
        async_wait0();
        __syncthreads();

        // --- scores: S = Q K^T ; B[d][key] built from kbuf ---
        Frag16 kb00, kb01, kb10, kb11;
#pragma unroll
        for (int g = 0; g < 8; ++g) {
            const int dk = ((g >> 2) << 4) + (h << 3) + ((g & 3) << 1);
            kb00.u[g] = kbu[((n)      * DHEAD + dk)      >> 1];
            kb01.u[g] = kbu[((n)      * DHEAD + 32 + dk) >> 1];
            kb10.u[g] = kbu[((16 + n) * DHEAD + dk)      >> 1];
            kb11.u[g] = kbu[((16 + n) * DHEAD + 32 + dk) >> 1];
        }
        v8f s0v = {}, s1v = {};
        s0v = wmma_f16(qa0.v, kb00.v, s0v);
        s0v = wmma_f16(qa1.v, kb01.v, s0v);
        s1v = wmma_f16(qa0.v, kb10.v, s1v);
        s1v = wmma_f16(qa1.v, kb11.v, s1v);

        // --- scale, mask, chunk row-max ---
        float t0[8], t1[8], cm[8];
#pragma unroll
        for (int r = 0; r < 8; ++r) {
            float a = s0v[r] * 0.125f;             // 1/sqrt(64)
            float c = s1v[r] * 0.125f;
            const size_t moff = (size_t)(r + 8 * h) * SEQ + kc;
            if (mp[moff + n])      a = -__builtin_inff();
            if (mp[moff + 16 + n]) c = -__builtin_inff();
            t0[r] = a; t1[r] = c;
            float m = fmaxf(a, c);
#pragma unroll
            for (int off = 1; off < 16; off <<= 1)
                m = fmaxf(m, __shfl_xor(m, off, 32));
            cm[r] = m;
        }

        // --- online softmax update + stage P into LDS (f16) ---
#pragma unroll
        for (int r = 0; r < 8; ++r) {
            const float mn    = fmaxf(mrow[r], cm[r]);
            const float alpha = __expf(mrow[r] - mn);   // NaN if row fully masked -> srow NaN -> out 0
            mrow[r] = mn;
            const float p0 = __expf(t0[r] - mn);
            const float p1 = __expf(t1[r] - mn);
            float ps = p0 + p1;
#pragma unroll
            for (int off = 1; off < 16; off <<= 1)
                ps += __shfl_xor(ps, off, 32);
            srow[r] = srow[r] * alpha + ps;
            o0[r] *= alpha; o1[r] *= alpha; o2[r] *= alpha; o3[r] *= alpha;
            pbuf[wave][r + 8 * h][n]      = (_Float16)p0;
            pbuf[wave][r + 8 * h][16 + n] = (_Float16)p1;
        }

        // --- reload P in A-fragment layout ---
        Frag16 pa;
#pragma unroll
        for (int g = 0; g < 8; ++g) {
            const int kk = ((g >> 2) << 4) + (h << 3) + ((g & 3) << 1);
            pa.u[g] = pw[(n * 32 + kk) >> 1];
        }

        // --- O += P V ; V^T[d][key] from vbuf ---
        Frag16 vb0, vb1, vb2, vb3;
#pragma unroll
        for (int g = 0; g < 8; ++g) {
            const int key = ((g >> 2) << 4) + (h << 3) + ((g & 3) << 1);
            vb0.u[g] = vbu[((0 * 16 + n) * 32 + key) >> 1];
            vb1.u[g] = vbu[((1 * 16 + n) * 32 + key) >> 1];
            vb2.u[g] = vbu[((2 * 16 + n) * 32 + key) >> 1];
            vb3.u[g] = vbu[((3 * 16 + n) * 32 + key) >> 1];
        }
        o0 = wmma_f16(pa.v, vb0.v, o0);
        o1 = wmma_f16(pa.v, vb1.v, o1);
        o2 = wmma_f16(pa.v, vb2.v, o2);
        o3 = wmma_f16(pa.v, vb3.v, o3);

        __syncthreads();   // protect kbuf/vbuf before next chunk's staging
    }

    // --- epilogue: normalize (NaN-safe like reference nan_to_num) and store ---
#pragma unroll
    for (int r = 0; r < 8; ++r) {
        const int tok = q0 + r + 8 * h;
        const size_t base = ((size_t)b * SEQ + tok) * DMODEL + hd * DHEAD;
        const bool ok = (srow[r] > 0.f);
        const float inv = ok ? (1.f / srow[r]) : 0.f;
        ctx[base + 0 * 16 + n] = (_Float16)(ok ? o0[r] * inv : 0.f);
        ctx[base + 1 * 16 + n] = (_Float16)(ok ? o1[r] * inv : 0.f);
        ctx[base + 2 * 16 + n] = (_Float16)(ok ? o2[r] * inv : 0.f);
        ctx[base + 3 * 16 + n] = (_Float16)(ok ? o3[r] * inv : 0.f);
    }
}

// ---------------------------------------------------------------------------
extern "C" void kernel_launch(void* const* d_in, const int* in_sizes, int n_in,
                              void* d_out, int out_size, void* d_ws, size_t ws_size,
                              hipStream_t stream) {
    (void)in_sizes; (void)n_in; (void)out_size; (void)ws_size;
    const float* Query = (const float*)d_in[0];
    const float* Key   = (const float*)d_in[1];
    const float* Value = (const float*)d_in[2];
    const unsigned char* maskp = (const unsigned char*)d_in[3];
    const float* WQ_w = (const float*)d_in[4];
    const float* WQ_b = (const float*)d_in[5];
    const float* WK_w = (const float*)d_in[6];
    const float* WK_b = (const float*)d_in[7];
    const float* WV_w = (const float*)d_in[8];
    const float* WV_b = (const float*)d_in[9];
    const float* WO_w = (const float*)d_in[10];
    const float* WO_b = (const float*)d_in[11];

    const size_t NE = (size_t)TOK * DMODEL;        // 6,291,456 f16 elems per buffer
    _Float16* q16  = (_Float16*)d_ws;              // [B,H,L,Dh]
    _Float16* k16  = q16 + NE;                     // [B,H,L,Dh]
    _Float16* vT16 = k16 + NE;                     // [B,H,Dh,L]
    _Float16* ctx  = vT16 + NE;                    // [B,L,D]

    const dim3 blk(256);
    const dim3 grd((TOK / 16) * (DMODEL / 16) / 8);   // 3072 blocks, 1 tile/wave

    proj_gemm<0, false><<<grd, blk, 0, stream>>>(Query, WQ_w, WQ_b, q16,  nullptr);
    proj_gemm<0, false><<<grd, blk, 0, stream>>>(Key,   WK_w, WK_b, k16,  nullptr);
    proj_gemm<1, false><<<grd, blk, 0, stream>>>(Value, WV_w, WV_b, vT16, nullptr);

    attn_kernel<<<dim3(BATCH * NHEADS * (SEQ / 64)), dim3(128), 0, stream>>>(
        q16, k16, vT16, maskp, ctx);

    proj_gemm<2, true><<<grd, blk, 0, stream>>>(ctx, WO_w, WO_b, nullptr, (float*)d_out);
}
